// SparseMoE_52055003627788
// MI455X (gfx1250) — matrix operand order repaired and checked by reading, not compiled, not run on hardware
//
#include <hip/hip_runtime.h>
#include <hip/hip_bf16.h>

// ---------------------------------------------------------------------------
// Sparse MoE (top-2 of 8 experts) for gfx1250 / MI455X.
// N=2048 tokens, D_IN=1024, D_HID=4096, D_OUT=1024, E=8, K=2.
//
// Pipeline (one stream):
//   0) zero d_out + expert counters
//   1) convert W1,W2 f32 -> bf16 in d_ws (one pass; 134MB bf16 fits 192MB L2)
//   2) gate: wave32/token, top-2 + softmax, append to per-expert lists
//   3) grouped FFN: 32-token tiles per block, v_wmma_f32_16x16x32_bf16.
//      K-loop is double-buffered (A prefetched one kc ahead, B one tile
//      ahead) and loop-interchanged so one A fragment feeds 4 (GEMM1) or
//      8 (GEMM2) WMMAs. D_HID staged 2x2048 through LDS; persistent y
//      accumulators; f32 atomic scatter-add combines K=2 contributions.
// ---------------------------------------------------------------------------

#define N_TOK  2048
#define D_IN   1024
#define D_HID  4096
#define D_OUT  1024
#define NEXP   8
#define MT     32              // token tile per workgroup
#define HSTG   2048            // hidden-dim stage width
#define XA_LD  (D_IN + 8)      // padded LDS leading dims (stride % 64 dwords == 4)
#define HS_LD  (HSTG + 8)

typedef __attribute__((ext_vector_type(16))) __bf16 bf16x16;
typedef __attribute__((ext_vector_type(8)))  float  f32x8;

union Frag { unsigned int u[8]; bf16x16 v; };

__device__ __forceinline__ unsigned int pk_bf16(float a, float b) {
    unsigned int ua = __builtin_bit_cast(unsigned int, a);
    unsigned int ub = __builtin_bit_cast(unsigned int, b);
    ua = (ua + 0x7FFFu + ((ua >> 16) & 1u)) >> 16;      // RNE
    ub = (ub + 0x7FFFu + ((ub >> 16) & 1u)) >> 16;
    return (ua & 0xFFFFu) | (ub << 16);
}

__device__ __forceinline__ __bf16 f2bf(float f) {
    unsigned int u = __builtin_bit_cast(unsigned int, f);
    u = (u + 0x7FFFu + ((u >> 16) & 1u)) >> 16;
    unsigned short s = (unsigned short)u;
    return __builtin_bit_cast(__bf16, s);
}

// A fragment (16x32 MxK bf16, ISA 7.12.2): per-lane dwords {0..3, 8..11} of a
// 16-dword K-chunk row; rows/offsets are 16B aligned -> two ds_load_b128.
__device__ __forceinline__ void loadA(Frag& f, const unsigned int* rowbase, int kc) {
    const uint4* p = (const uint4*)(rowbase + kc * 16);
    uint4 q0 = p[0];                         // dwords 0..3
    uint4 q1 = p[2];                         // dwords 8..11
    f.u[0] = q0.x; f.u[1] = q0.y; f.u[2] = q0.z; f.u[3] = q0.w;
    f.u[4] = q1.x; f.u[5] = q1.y; f.u[6] = q1.z; f.u[7] = q1.w;
}

// B fragment (32x16 KxN bf16): two 8-bf16 (16B) runs from the bf16 weight
// row at K offsets kc*32 + 8*(lane>=16) and +16 -> two global_load_b128.
__device__ __forceinline__ void loadB(Frag& f, const __bf16* wrow, int kc, int half) {
    const uint4* p = (const uint4*)(wrow + kc * 32 + half * 8);
    uint4 q0 = p[0], q1 = p[2];
    f.u[0] = q0.x; f.u[1] = q0.y; f.u[2] = q0.z; f.u[3] = q0.w;
    f.u[4] = q1.x; f.u[5] = q1.y; f.u[6] = q1.z; f.u[7] = q1.w;
}

// ---------------------------------------------------------------------------
// Kernel 0: zero output (atomics accumulate) + expert counters.
// ---------------------------------------------------------------------------
__global__ void moe_init_kernel(float4* __restrict__ out4, int* __restrict__ cnt) {
    long i = (long)blockIdx.x * blockDim.x + threadIdx.x;
    const long n4 = (long)N_TOK * D_OUT / 4;
    if (i < n4) out4[i] = make_float4(0.f, 0.f, 0.f, 0.f);
    if (i < NEXP) cnt[i] = 0;
}

// ---------------------------------------------------------------------------
// Kernel 1: f32 -> bf16 weight conversion (float4 in, uint2 out).
// ---------------------------------------------------------------------------
__global__ void moe_cvt_kernel(const float4* __restrict__ src,
                               uint2* __restrict__ dst, long n4) {
    long i = (long)blockIdx.x * blockDim.x + threadIdx.x;
    if (i < n4) {
        float4 f = src[i];
        dst[i] = make_uint2(pk_bf16(f.x, f.y), pk_bf16(f.z, f.w));
    }
}

// ---------------------------------------------------------------------------
// Kernel 2: gate. One wave32 per token.
// ---------------------------------------------------------------------------
__global__ __launch_bounds__(256) void moe_gate_kernel(
        const float* __restrict__ x, const float* __restrict__ gw,
        const float* __restrict__ gb, int* __restrict__ cnt,
        int* __restrict__ tokList, float* __restrict__ wList) {
    const int wave = threadIdx.x >> 5;
    const int lane = threadIdx.x & 31;
    const int t = blockIdx.x * (blockDim.x >> 5) + wave;
    if (t >= N_TOK) return;

    float p[NEXP];
    #pragma unroll
    for (int e = 0; e < NEXP; ++e) p[e] = 0.f;

    const float* xr = x + (long)t * D_IN;
    for (int i = lane; i < D_IN; i += 32) {
        float xv = xr[i];
        #pragma unroll
        for (int e = 0; e < NEXP; ++e) p[e] += xv * gw[e * D_IN + i];
    }
    #pragma unroll
    for (int e = 0; e < NEXP; ++e) {
        #pragma unroll
        for (int s = 16; s > 0; s >>= 1) p[e] += __shfl_xor(p[e], s, 32);
    }
    if (lane == 0) {
        #pragma unroll
        for (int e = 0; e < NEXP; ++e) p[e] += gb[e];
        int i0 = 0;
        #pragma unroll
        for (int e = 1; e < NEXP; ++e) if (p[e] > p[i0]) i0 = e;
        int i1 = -1;
        #pragma unroll
        for (int e = 0; e < NEXP; ++e)
            if (e != i0 && (i1 < 0 || p[e] > p[i1])) i1 = e;
        float e1 = __expf(p[i1] - p[i0]);            // stable 2-way softmax
        float inv = 1.f / (1.f + e1);
        int pos0 = atomicAdd(&cnt[i0], 1);
        tokList[i0 * N_TOK + pos0] = t;  wList[i0 * N_TOK + pos0] = inv;
        int pos1 = atomicAdd(&cnt[i1], 1);
        tokList[i1 * N_TOK + pos1] = t;  wList[i1 * N_TOK + pos1] = e1 * inv;
    }
}

// ---------------------------------------------------------------------------
// Kernel 3: grouped expert FFN.
// grid = NEXP * (N_TOK/MT) = 512; block = 512 (16 waves = 2 Mtiles x 8 cgrps).
// LDS: xA[32][1032] bf16 (66KB) | hS[32][2056] bf16 (128.5KB) = 193KB dynamic.
// C/D fragment (f32 16x16): VGPR r -> M = r + 8*(lane>=16), N = lane&15.
// ---------------------------------------------------------------------------
#define SM_XA   0
#define SM_HS   (MT * XA_LD * 2)                 // 66048
#define SM_SIZE (SM_HS + MT * HS_LD * 2)         // 197632

__global__ __launch_bounds__(512) void moe_ffn_kernel(
        const float* __restrict__ x,
        const __bf16* __restrict__ W1b, const float* __restrict__ b1,
        const __bf16* __restrict__ W2b, const float* __restrict__ b2,
        float* __restrict__ out,
        const int* __restrict__ cnt,  const int* __restrict__ tokList,
        const float* __restrict__ wList) {
    extern __shared__ char smem[];
    __bf16* xA = (__bf16*)(smem + SM_XA);
    __bf16* hS = (__bf16*)(smem + SM_HS);

    const int e    = blockIdx.x >> 6;            // 64 blocks per expert
    const int blk  = blockIdx.x & 63;
    const int cntE = cnt[e];
    const int base = blk * MT;
    if (base >= cntE) return;                    // uniform early-exit

    const int tid = threadIdx.x;

    // ---- gather x rows -> xA (bf16). 16 threads/row, 64 f32 each. ----
    {
        int r  = tid >> 4;                       // 0..31
        int c0 = (tid & 15) * 64;
        int t  = (base + r < cntE) ? tokList[e * N_TOK + base + r] : -1;
        const float* src = x + (long)(t < 0 ? 0 : t) * D_IN + c0;
        unsigned int* dst = (unsigned int*)(xA + r * XA_LD + c0);
        #pragma unroll 8
        for (int i = 0; i < 64; i += 2) {
            float a = (t >= 0) ? src[i]     : 0.f;
            float b = (t >= 0) ? src[i + 1] : 0.f;
            dst[i >> 1] = pk_bf16(a, b);
        }
    }
    __syncthreads();

    const int wave = tid >> 5;
    const int lane = tid & 31;
    const int half = lane >> 4;
    const int l15  = lane & 15;
    const int mgrp = wave & 1;                   // which 16-token row tile
    const int cgrp = wave >> 1;                  // 0..7 column group

    const __bf16* W1e = W1b + (long)e * D_HID * D_IN;
    const __bf16* W2e = W2b + (long)e * D_OUT * D_HID;
    const float*  b1e = b1 + e * D_HID;
    const float*  b2e = b2 + e * D_OUT;

    const unsigned int* xrow =
        (const unsigned int*)(xA + (mgrp * 16 + l15) * XA_LD) + half * 4;
    const unsigned int* hrow =
        (const unsigned int*)(hS + (mgrp * 16 + l15) * HS_LD) + half * 4;

    f32x8 yacc[8];
    #pragma unroll
    for (int i = 0; i < 8; ++i)
        yacc[i] = (f32x8){0.f, 0.f, 0.f, 0.f, 0.f, 0.f, 0.f, 0.f};

    for (int s = 0; s < D_HID / HSTG; ++s) {     // 2 hidden stages
        // ---- GEMM1: hS[32][2048] = relu(xA @ W1[e][stage]^T + b1) ----
        // 16 tiles/wave in 4 groups of 4; one A fragment feeds 4 WMMAs.
        for (int g = 0; g < 4; ++g) {
            const __bf16* wr[4];
            #pragma unroll
            for (int j = 0; j < 4; ++j) {
                int hid = s * HSTG + (cgrp + 8 * (4 * g + j)) * 16 + l15;
                wr[j] = W1e + (long)hid * D_IN;
            }
            f32x8 acc[4];
            #pragma unroll
            for (int j = 0; j < 4; ++j)
                acc[j] = (f32x8){0.f, 0.f, 0.f, 0.f, 0.f, 0.f, 0.f, 0.f};

            const int KC = D_IN / 32;
            Frag a, an, b, bn;
            loadA(a, xrow, 0);
            loadB(b, wr[0], 0, half);
            for (int kc = 0; kc < KC; ++kc) {
                const int kn = (kc + 1 < KC) ? kc + 1 : kc;
                loadA(an, xrow, kn);             // A prefetch, one kc ahead
                #pragma unroll
                for (int j = 0; j < 4; ++j) {    // B prefetch, one tile ahead
                    if (j < 3) loadB(bn, wr[j + 1], kc, half);
                    else       loadB(bn, wr[0], kn, half);
                    acc[j] = __builtin_amdgcn_wmma_f32_16x16x32_bf16(
                            false, a.v, false, b.v, (short)0, acc[j], false, false);
                    b = bn;
                }
                a = an;
            }
            #pragma unroll
            for (int j = 0; j < 4; ++j) {
                int ht  = cgrp + 8 * (4 * g + j);
                int hid = s * HSTG + ht * 16 + l15;
                float bias = b1e[hid];
                #pragma unroll
                for (int r = 0; r < 8; ++r) {
                    float v = fmaxf(acc[j][r] + bias, 0.f);  // bias + ReLU
                    int m = mgrp * 16 + half * 8 + r;
                    hS[m * HS_LD + ht * 16 + l15] = f2bf(v);
                }
            }
        }
        __syncthreads();

        // ---- GEMM2 partial: yacc += hS @ W2[e][:, stage]^T ----
        // One A fragment feeds all 8 output tiles (persistent accumulators).
        {
            const __bf16* wr[8];
            #pragma unroll
            for (int j = 0; j < 8; ++j) {
                int col = (cgrp + 8 * j) * 16 + l15;
                wr[j] = W2e + (long)col * D_HID + s * HSTG;
            }
            const int KC = HSTG / 32;
            Frag a, an, b, bn;
            loadA(a, hrow, 0);
            loadB(b, wr[0], 0, half);
            for (int kc = 0; kc < KC; ++kc) {
                const int kn = (kc + 1 < KC) ? kc + 1 : kc;
                loadA(an, hrow, kn);
                #pragma unroll
                for (int j = 0; j < 8; ++j) {
                    if (j < 7) loadB(bn, wr[j + 1], kc, half);
                    else       loadB(bn, wr[0], kn, half);
                    yacc[j] = __builtin_amdgcn_wmma_f32_16x16x32_bf16(
                            false, a.v, false, b.v, (short)0, yacc[j], false, false);
                    b = bn;
                }
                a = an;
            }
        }
        __syncthreads();                         // hS reused next stage
    }

    // ---- epilogue: out[t] += gate_w * (y + b2) (f32 atomics) ----
    int   tki[8];
    float twv[8];
    #pragma unroll
    for (int r = 0; r < 8; ++r) {
        int idx = base + mgrp * 16 + half * 8 + r;
        bool ok = idx < cntE;
        tki[r] = ok ? tokList[e * N_TOK + idx] : -1;
        twv[r] = ok ? wList[e * N_TOK + idx] : 0.f;
    }
    #pragma unroll
    for (int i = 0; i < 8; ++i) {
        const int col = (cgrp + 8 * i) * 16 + l15;
        float bias = b2e[col];
        #pragma unroll
        for (int r = 0; r < 8; ++r) {
            if (tki[r] >= 0)
                atomicAdd(out + (long)tki[r] * D_OUT + col,
                          twv[r] * (yacc[i][r] + bias));
        }
    }
}

// ---------------------------------------------------------------------------
// Launcher. Inputs: x, gate_w, gate_b, W1, b1, W2, b2 (f32). Output f32.
// ws: cnt[8]@0 | tokList 64KB@256 | wList 64KB | W1bf 64MB | W2bf 64MB (~128MB)
// ---------------------------------------------------------------------------
extern "C" void kernel_launch(void* const* d_in, const int* in_sizes, int n_in,
                              void* d_out, int out_size, void* d_ws, size_t ws_size,
                              hipStream_t stream) {
    const float* x  = (const float*)d_in[0];
    const float* gw = (const float*)d_in[1];
    const float* gb = (const float*)d_in[2];
    const float* W1 = (const float*)d_in[3];
    const float* b1 = (const float*)d_in[4];
    const float* W2 = (const float*)d_in[5];
    const float* b2 = (const float*)d_in[6];
    float* out = (float*)d_out;

    char* ws = (char*)d_ws;
    int*    cnt     = (int*)ws;
    int*    tokList = (int*)(ws + 256);
    float*  wList   = (float*)(ws + 256 + (size_t)NEXP * N_TOK * 4);
    __bf16* W1bf    = (__bf16*)(ws + 131328);                 // 256-aligned
    __bf16* W2bf    = (__bf16*)(ws + 131328 + (size_t)NEXP * D_HID * D_IN * 2);

    // 0) zero output + counters
    {
        const long n4 = (long)N_TOK * D_OUT / 4;
        moe_init_kernel<<<(int)((n4 + 255) / 256), 256, 0, stream>>>(
            (float4*)out, cnt);
    }
    // 1) weight conversion f32 -> bf16 (bf16 weights stay resident in 192MB L2)
    {
        const long n4 = (long)NEXP * D_HID * D_IN / 4;        // 8.39M float4
        moe_cvt_kernel<<<(int)((n4 + 255) / 256), 256, 0, stream>>>(
            (const float4*)W1, (uint2*)W1bf, n4);
        moe_cvt_kernel<<<(int)((n4 + 255) / 256), 256, 0, stream>>>(
            (const float4*)W2, (uint2*)W2bf, n4);
    }
    // 2) gate + routing (8 wave32 tokens per 256-thread block)
    moe_gate_kernel<<<N_TOK / 8, 256, 0, stream>>>(x, gw, gb, cnt, tokList, wList);
    // 3) grouped expert FFN
    moe_ffn_kernel<<<NEXP * (N_TOK / MT), 512, SM_SIZE, stream>>>(
        x, W1bf, b1, W2bf, b2, out, cnt, tokList, wList);
}
